// LSTM1_18399639896917
// MI455X (gfx1250) — matrix operand order repaired and checked by reading, not compiled, not run on hardware
//
#include <hip/hip_runtime.h>
#include <stdint.h>

// Problem constants (from reference)
#define H_DIM   2048
#define G4H     8192      // 4*H
#define N_INP   512
#define N_OUT   512
#define T_STEPS 4096

// Scan kernel configuration
#define SCAN_BLOCKS 512
#define TPB         256
#define UPB   (H_DIM / SCAN_BLOCKS)   // 4 hidden units per block
#define ROWS  (4 * UPB)               // 16 matrix rows per block per layer
#define NWAVE (TPB / 32)              // 8 waves
#define RPW   (ROWS / NWAVE)          // 2 rows per wave

typedef unsigned short bf16s;

// ---------- bf16 helpers (stored as raw u16, fp32 math) ----------
__device__ __forceinline__ float bflo(unsigned u) { return __uint_as_float(u << 16); }
__device__ __forceinline__ float bfhi(unsigned u) { return __uint_as_float(u & 0xFFFF0000u); }
__device__ __forceinline__ float bfs2f(bf16s v)   { return __uint_as_float(((unsigned)v) << 16); }
__device__ __forceinline__ bf16s f2bf(float f) {
  unsigned u = __float_as_uint(f);
  u += 0x7FFFu + ((u >> 16) & 1u);     // round-to-nearest-even
  return (bf16s)(u >> 16);
}
__device__ __forceinline__ float dot8(uint4 w, uint4 h) {
  float s = bflo(w.x) * bflo(h.x);
  s = fmaf(bfhi(w.x), bfhi(h.x), s);
  s = fmaf(bflo(w.y), bflo(h.y), s);
  s = fmaf(bfhi(w.y), bfhi(h.y), s);
  s = fmaf(bflo(w.z), bflo(h.z), s);
  s = fmaf(bfhi(w.z), bfhi(h.z), s);
  s = fmaf(bflo(w.w), bflo(h.w), s);
  s = fmaf(bfhi(w.w), bfhi(h.w), s);
  return s;
}
__device__ __forceinline__ float sigm(float x) { return 1.0f / (1.0f + __expf(-x)); }

// ---------- grid-wide sense-reversing barrier (L2 atomics) ----------
__device__ __forceinline__ void gbar(unsigned* cnt, unsigned* gen, unsigned nb) {
  __syncthreads();
  if (threadIdx.x == 0) {
    __threadfence();
    unsigned g = __hip_atomic_load(gen, __ATOMIC_RELAXED, __HIP_MEMORY_SCOPE_AGENT);
    unsigned a = __hip_atomic_fetch_add(cnt, 1u, __ATOMIC_ACQ_REL, __HIP_MEMORY_SCOPE_AGENT);
    if (a == nb - 1u) {
      __hip_atomic_store(cnt, 0u, __ATOMIC_RELAXED, __HIP_MEMORY_SCOPE_AGENT);
      __hip_atomic_fetch_add(gen, 1u, __ATOMIC_RELEASE, __HIP_MEMORY_SCOPE_AGENT);
    } else {
      while (__hip_atomic_load(gen, __ATOMIC_ACQUIRE, __HIP_MEMORY_SCOPE_AGENT) == g)
        __builtin_amdgcn_s_sleep(1);
    }
  }
  __syncthreads();
}

// ---------- WMMA fragment types ----------
typedef __bf16 v16bf __attribute__((ext_vector_type(16)));
typedef float  v8f   __attribute__((ext_vector_type(8)));
union AF { v16bf v; unsigned u[8]; };
union CF { v8f  v; float    f[8]; };

// =====================================================================
// NT GEMM (both operands K-contiguous): C[m,n] = sum_k A[m,k]*B[n,k] + bias[n]
// A: [M,K] bf16 row-major; B: [N,K] bf16 row-major; out bf16 or f32.
// Block = 8 waves, each wave owns one 16x16 C tile; block tile = 32x64.
// Fragment layouts per CDNA5 ISA 7.12.2 (16-bit A 16x32, B 32x16, f32 C 16x16).
// =====================================================================
__global__ void __launch_bounds__(TPB)
k_gemm_nt_bf16(const bf16s* __restrict__ A, const bf16s* __restrict__ B,
               const float* __restrict__ bias, void* __restrict__ Cout,
               int M, int N, int K, int outIsBf16)
{
  const int lane = threadIdx.x & 31;
  const int wave = threadIdx.x >> 5;
  const int wr = wave >> 2, wc = wave & 3;            // 2x4 wave grid
  const int m0 = blockIdx.y * 32 + wr * 16;
  const int n0 = blockIdx.x * 64 + wc * 16;
  const int ml = lane & 15;
  const int hi = lane >> 4;

  const bf16s* arow = A + (size_t)(m0 + ml) * K;
  const bf16s* brow = B + (size_t)(n0 + ml) * K;

  AF a, b;
  CF c;
#pragma unroll
  for (int r = 0; r < 8; ++r) c.f[r] = 0.0f;

  for (int k0 = 0; k0 < K; k0 += 32) {
#pragma unroll
    for (int j = 0; j < 8; ++j) {
      // A 16x32: low lanes K 0-7 / 16-23, high lanes +8 (packed K pairs)
      int ka = k0 + ((j < 4) ? (j * 2) : (16 + (j - 4) * 2)) + hi * 8;
      a.u[j] = *(const unsigned*)(arow + ka);
      // B 32x16: low lanes K 0-7 / 16-23, high lanes +8 (packed K pairs)
      int kb = k0 + ((j < 4) ? (j * 2) : (16 + (j - 4) * 2)) + hi * 8;
      b.u[j] = *(const unsigned*)(brow + kb);
    }
    c.v = __builtin_amdgcn_wmma_f32_16x16x32_bf16(false, a.v, false, b.v,
                                                  (short)0, c.v, false, false);
  }

#pragma unroll
  for (int r = 0; r < 8; ++r) {
    int m = m0 + hi * 8 + r;      // C layout: VGPR r -> M=r (lanes 0-15) / M=8+r
    int n = n0 + ml;
    float v = c.f[r] + (bias ? bias[n] : 0.0f);
    if (outIsBf16) ((bf16s*)Cout)[(size_t)m * N + n] = f2bf(v);
    else           ((float*)Cout)[(size_t)m * N + n] = v;
  }
}

// =====================================================================
// Persistent two-layer LSTM scan. 512 blocks, each owns 4 hidden units.
// bf16 weights (100.7 MB total -> L2 resident), fp32 accumulate,
// cell state in registers, 2 grid barriers per timestep.
// =====================================================================
__global__ void __launch_bounds__(TPB)
k_lstm_scan(const bf16s* __restrict__ Whh0, const bf16s* __restrict__ Wih1,
            const bf16s* __restrict__ Whh1, const bf16s* __restrict__ xproj,
            const float* __restrict__ b1sum,
            bf16s* hb0, bf16s* hb1, bf16s* __restrict__ h1seq,
            unsigned* bar, float* __restrict__ outHidden)
{
  __shared__ uint4 sh0q[H_DIM * 2 / 16];   // h0 as bf16, 4KB
  __shared__ uint4 sh1q[H_DIM * 2 / 16];   // h1 as bf16, 4KB
  __shared__ float spart[ROWS];

  const int tid  = threadIdx.x;
  const int blk  = blockIdx.x;
  const int wave = tid >> 5, lane = tid & 31;
  unsigned* cnt = bar;
  unsigned* gen = bar + 1;

  float c0 = 0.0f, c1 = 0.0f, h0f = 0.0f, h1f = 0.0f;  // owner-thread state (tid < UPB)

  // preload h0 (zeros) — phase B of each step refreshes sh0q for the next step
  sh0q[tid] = ((const uint4*)hb0)[tid];
  __syncthreads();

  for (int t = 0; t < T_STEPS; ++t) {
    // ---- Phase A: y0 = W_hh_l0 @ h0_{t-1} for this block's 16 rows, then gates0 ----
#pragma unroll
    for (int r = 0; r < RPW; ++r) {
      int ridx = wave * RPW + r;                 // 0..15 = gate*UPB + unit
      int g = ridx / UPB, ul = ridx % UPB;
      size_t row = (size_t)g * H_DIM + (size_t)blk * UPB + ul;
      const uint4* wrow = (const uint4*)(Whh0 + row * H_DIM);
      float acc = 0.0f;
#pragma unroll
      for (int it = 0; it < H_DIM / 256; ++it) { // 8 iters of b128 per lane
        int idx = it * 32 + lane;
        acc += dot8(wrow[idx], sh0q[idx]);
      }
#pragma unroll
      for (int off = 16; off; off >>= 1) acc += __shfl_xor(acc, off);
      if (lane == 0) spart[ridx] = acc;
    }
    __syncthreads();
    if (tid < UPB) {
      int u = blk * UPB + tid;
      const bf16s* xp = xproj + (size_t)t * G4H;
      float gi = spart[0 * UPB + tid] + bfs2f(xp[u]);
      float gf = spart[1 * UPB + tid] + bfs2f(xp[H_DIM + u]);
      float gg = spart[2 * UPB + tid] + bfs2f(xp[2 * H_DIM + u]);
      float go = spart[3 * UPB + tid] + bfs2f(xp[3 * H_DIM + u]);
      gi = sigm(gi); gf = sigm(gf); gg = tanhf(gg); go = sigm(go);
      c0  = gf * c0 + gi * gg;
      h0f = go * tanhf(c0);
      hb0[u] = f2bf(h0f);
    }
    gbar(cnt, gen, SCAN_BLOCKS);

    // ---- Phase B: y1 = W_ih_l1 @ h0_t + W_hh_l1 @ h1_{t-1}, then gates1 ----
    sh0q[tid] = ((const uint4*)hb0)[tid];        // h0_t (also reused by next step's phase A)
    sh1q[tid] = ((const uint4*)hb1)[tid];        // h1_{t-1}
    __syncthreads();
#pragma unroll
    for (int r = 0; r < RPW; ++r) {
      int ridx = wave * RPW + r;
      int g = ridx / UPB, ul = ridx % UPB;
      size_t row = (size_t)g * H_DIM + (size_t)blk * UPB + ul;
      const uint4* w1 = (const uint4*)(Wih1 + row * H_DIM);
      const uint4* w2 = (const uint4*)(Whh1 + row * H_DIM);
      float acc = 0.0f;
#pragma unroll
      for (int it = 0; it < H_DIM / 256; ++it) {
        int idx = it * 32 + lane;
        acc += dot8(w1[idx], sh0q[idx]);
        acc += dot8(w2[idx], sh1q[idx]);
      }
#pragma unroll
      for (int off = 16; off; off >>= 1) acc += __shfl_xor(acc, off);
      if (lane == 0) spart[ridx] = acc;
    }
    __syncthreads();
    if (tid < UPB) {
      int u = blk * UPB + tid;
      float gi = spart[0 * UPB + tid] + b1sum[u];
      float gf = spart[1 * UPB + tid] + b1sum[H_DIM + u];
      float gg = spart[2 * UPB + tid] + b1sum[2 * H_DIM + u];
      float go = spart[3 * UPB + tid] + b1sum[3 * H_DIM + u];
      gi = sigm(gi); gf = sigm(gf); gg = tanhf(gg); go = sigm(go);
      c1  = gf * c1 + gi * gg;
      h1f = go * tanhf(c1);
      bf16s hv = f2bf(h1f);
      hb1[u] = hv;
      h1seq[(size_t)t * H_DIM + u] = hv;         // spill for batched pi / v heads
    }
    gbar(cnt, gen, SCAN_BLOCKS);
  }

  if (tid < UPB) {                               // final hidden/cell state (fp32)
    int u = blk * UPB + tid;
    outHidden[u]              = h0f;  // hidden_h[0]
    outHidden[H_DIM + u]      = h1f;  // hidden_h[1]
    outHidden[2 * H_DIM + u]  = c0;   // hidden_c[0]
    outHidden[3 * H_DIM + u]  = c1;   // hidden_c[1]
  }
}

// ---------- small utility kernels ----------
__global__ void k_f2bf(const float* __restrict__ s, bf16s* __restrict__ d, long n) {
  long i = (long)blockIdx.x * blockDim.x + threadIdx.x;
  long st = (long)gridDim.x * blockDim.x;
  for (; i < n; i += st) d[i] = f2bf(s[i]);
}
__global__ void k_addvec(const float* __restrict__ a, const float* __restrict__ b,
                         float* __restrict__ d, int n) {
  int i = blockIdx.x * blockDim.x + threadIdx.x;
  if (i < n) d[i] = a[i] + b[i];
}
__global__ void k_init(unsigned* hb0, unsigned* hb1, unsigned* bar) {
  int i = blockIdx.x * blockDim.x + threadIdx.x;
  if (i < H_DIM / 2) { hb0[i] = 0u; hb1[i] = 0u; }
  if (i < 2) bar[i] = 0u;
}

// v[t] = dot(h1seq[t], W_V) + b_V ; one wave per timestep
__global__ void __launch_bounds__(TPB)
k_vhead(const bf16s* __restrict__ h1seq, const float* __restrict__ WV,
        const float* __restrict__ bV, float* __restrict__ out)
{
  int lane = threadIdx.x & 31;
  int t = blockIdx.x * NWAVE + (threadIdx.x >> 5);
  const uint4*  hq = (const uint4*)(h1seq + (size_t)t * H_DIM);
  const float4* wq = (const float4*)WV;
  float acc = 0.0f;
#pragma unroll
  for (int it = 0; it < H_DIM / 256; ++it) {
    int idx = it * 32 + lane;
    uint4  h  = hq[idx];
    float4 w0 = wq[idx * 2];
    float4 w1 = wq[idx * 2 + 1];
    acc += bflo(h.x) * w0.x + bfhi(h.x) * w0.y + bflo(h.y) * w0.z + bfhi(h.y) * w0.w
         + bflo(h.z) * w1.x + bfhi(h.z) * w1.y + bflo(h.w) * w1.z + bfhi(h.w) * w1.w;
  }
#pragma unroll
  for (int off = 16; off; off >>= 1) acc += __shfl_xor(acc, off);
  if (lane == 0) out[t] = acc + bV[0];
}

// in-place softmax over rows of 512 (one block per row)
__global__ void __launch_bounds__(TPB) k_softmax512(float* __restrict__ p) {
  __shared__ float red[TPB];
  float* row = p + (size_t)blockIdx.x * N_OUT;
  int tid = threadIdx.x;
  float x0 = row[tid * 2], x1 = row[tid * 2 + 1];
  red[tid] = fmaxf(x0, x1);
  __syncthreads();
  for (int s = 128; s > 0; s >>= 1) { if (tid < s) red[tid] = fmaxf(red[tid], red[tid + s]); __syncthreads(); }
  float M = red[0];
  __syncthreads();
  float e0 = __expf(x0 - M), e1 = __expf(x1 - M);
  red[tid] = e0 + e1;
  __syncthreads();
  for (int s = 128; s > 0; s >>= 1) { if (tid < s) red[tid] += red[tid + s]; __syncthreads(); }
  float inv = 1.0f / red[0];
  row[tid * 2] = e0 * inv;
  row[tid * 2 + 1] = e1 * inv;
}

// =====================================================================
extern "C" void kernel_launch(void* const* d_in, const int* in_sizes, int n_in,
                              void* d_out, int out_size, void* d_ws, size_t ws_size,
                              hipStream_t stream)
{
  (void)in_sizes; (void)n_in; (void)out_size; (void)ws_size;
  const float* inputs  = (const float*)d_in[0];
  const float* W_ih_l0 = (const float*)d_in[1];
  const float* W_hh_l0 = (const float*)d_in[2];
  const float* b_ih_l0 = (const float*)d_in[3];
  const float* b_hh_l0 = (const float*)d_in[4];
  const float* W_ih_l1 = (const float*)d_in[5];
  const float* W_hh_l1 = (const float*)d_in[6];
  const float* b_ih_l1 = (const float*)d_in[7];
  const float* b_hh_l1 = (const float*)d_in[8];
  const float* W_V     = (const float*)d_in[9];
  const float* b_V     = (const float*)d_in[10];
  const float* W_pi    = (const float*)d_in[11];
  const float* b_pi    = (const float*)d_in[12];

  // workspace layout (~199 MB; bf16 weight images fit MI455X's 192MB L2 for the scan)
  char* ws = (char*)d_ws;
  size_t off = 0;
  auto alloc = [&](size_t bytes) -> void* {
    void* p = ws + off;
    off = (off + bytes + 255) & ~(size_t)255;
    return p;
  };
  bf16s* Whh0b  = (bf16s*)alloc((size_t)G4H * H_DIM * 2);
  bf16s* Wih1b  = (bf16s*)alloc((size_t)G4H * H_DIM * 2);
  bf16s* Whh1b  = (bf16s*)alloc((size_t)G4H * H_DIM * 2);
  bf16s* Wih0b  = (bf16s*)alloc((size_t)G4H * N_INP * 2);
  bf16s* Xb     = (bf16s*)alloc((size_t)T_STEPS * N_INP * 2);
  bf16s* Wpib   = (bf16s*)alloc((size_t)N_OUT * H_DIM * 2);
  bf16s* xprojb = (bf16s*)alloc((size_t)T_STEPS * G4H * 2);
  bf16s* h1seqb = (bf16s*)alloc((size_t)T_STEPS * H_DIM * 2);
  float* b0sum  = (float*)alloc((size_t)G4H * 4);
  float* b1sum  = (float*)alloc((size_t)G4H * 4);
  bf16s* hb0    = (bf16s*)alloc((size_t)H_DIM * 2);
  bf16s* hb1    = (bf16s*)alloc((size_t)H_DIM * 2);
  unsigned* bar = (unsigned*)alloc(64);

  // 1) fp32 -> bf16 conversions
  k_f2bf<<<2048, TPB, 0, stream>>>(W_hh_l0, Whh0b, (long)G4H * H_DIM);
  k_f2bf<<<2048, TPB, 0, stream>>>(W_ih_l1, Wih1b, (long)G4H * H_DIM);
  k_f2bf<<<2048, TPB, 0, stream>>>(W_hh_l1, Whh1b, (long)G4H * H_DIM);
  k_f2bf<<<2048, TPB, 0, stream>>>(W_ih_l0, Wih0b, (long)G4H * N_INP);
  k_f2bf<<<2048, TPB, 0, stream>>>(inputs,  Xb,    (long)T_STEPS * N_INP);
  k_f2bf<<<2048, TPB, 0, stream>>>(W_pi,    Wpib,  (long)N_OUT * H_DIM);

  // 2) bias prep + state/barrier init
  k_addvec<<<G4H / TPB, TPB, 0, stream>>>(b_ih_l0, b_hh_l0, b0sum, G4H);
  k_addvec<<<G4H / TPB, TPB, 0, stream>>>(b_ih_l1, b_hh_l1, b1sum, G4H);
  k_init<<<8, TPB, 0, stream>>>((unsigned*)hb0, (unsigned*)hb1, bar);

  // 3) x_proj = inputs @ W_ih_l0^T + (b_ih_l0+b_hh_l0)   [WMMA bf16, out bf16]
  dim3 gx(G4H / 64, T_STEPS / 32);
  k_gemm_nt_bf16<<<gx, TPB, 0, stream>>>(Xb, Wih0b, b0sum, xprojb,
                                         T_STEPS, G4H, N_INP, 1);

  // 4) persistent recurrent scan
  float* outHidden = (float*)d_out + (T_STEPS + (size_t)T_STEPS * N_OUT);
  k_lstm_scan<<<SCAN_BLOCKS, TPB, 0, stream>>>(Whh0b, Wih1b, Whh1b, xprojb, b1sum,
                                               hb0, hb1, h1seqb, bar, outHidden);

  // 5) pi logits = h1_seq @ W_pi^T + b_pi (fp32, directly into d_out), then softmax
  float* piOut = (float*)d_out + T_STEPS;
  dim3 gp(N_OUT / 64, T_STEPS / 32);
  k_gemm_nt_bf16<<<gp, TPB, 0, stream>>>(h1seqb, Wpib, b_pi, piOut,
                                         T_STEPS, N_OUT, H_DIM, 0);
  k_softmax512<<<T_STEPS, TPB, 0, stream>>>(piOut);

  // 6) value head -> outputs[T]
  k_vhead<<<T_STEPS / NWAVE, TPB, 0, stream>>>(h1seqb, W_V, b_V, (float*)d_out);
}
